// Attention_21002390077599
// MI455X (gfx1250) — matrix operand order repaired and checked by reading, not compiled, not run on hardware
//
#include <hip/hip_runtime.h>

// ---------- problem constants ----------
constexpr int Bc = 4;
constexpr int Nc = 2048;
constexpr int Cc = 1024;
constexpr int Hc = 16;
constexpr int Dc = 64;          // head dim
constexpr float SCALE = 0.125f; // 64^-0.5

// ---------- types ----------
typedef __bf16 bf16;
typedef __attribute__((ext_vector_type(16))) __bf16 v16bf;
typedef __attribute__((ext_vector_type(8)))  float  v8f;
typedef __attribute__((ext_vector_type(4)))  unsigned int u32x4;

#define ZERO8 {0.f,0.f,0.f,0.f,0.f,0.f,0.f,0.f}

// Assemble a 16-element bf16 fragment from two 16-byte chunks.
// A-fragment (16x32, row = lane&15): elements 0..7 = K kb..kb+7 at p0,
// elements 8..15 = K kb+8..kb+15 at p1 = p0+8 (kb = 16*(lane>>4)).
// B-fragment (32x16, col = lane&15): 16 consecutive K values, p1 = p0+8
// (kb = 16*(lane>>4)).
__device__ __forceinline__ v16bf make_frag(const bf16* p0, const bf16* p1) {
    union { u32x4 q[2]; v16bf v; } u;
    u.q[0] = *reinterpret_cast<const u32x4*>(p0);
    u.q[1] = *reinterpret_cast<const u32x4*>(p1);
    return u.v;
}

__device__ __forceinline__ v8f wmma_bf16(v16bf a, v16bf b, v8f c) {
    // D = A(16x32 bf16) * B(32x16 bf16) + C(16x16 f32)
    return __builtin_amdgcn_wmma_f32_16x16x32_bf16(
        false, a, false, b, (short)0, c, false, false);
}

__device__ __forceinline__ float rmax16(float v) {
    #pragma unroll
    for (int m = 1; m <= 8; m <<= 1) v = fmaxf(v, __shfl_xor(v, m, 32));
    return v;
}
__device__ __forceinline__ float rsum16(float v) {
    #pragma unroll
    for (int m = 1; m <= 8; m <<= 1) v += __shfl_xor(v, m, 32);
    return v;
}

// Async DMA: 16 bytes global -> LDS, tracked by ASYNCcnt (no VGPR round-trip).
__device__ __forceinline__ void async_ld_b128(const bf16* g, const bf16* lds) {
    unsigned           loff = (unsigned)(unsigned long long)(uintptr_t)lds; // LDS offset = low 32 bits
    unsigned long long ga   = (unsigned long long)(uintptr_t)g;
    asm volatile("global_load_async_to_lds_b128 %0, %1, off"
                 :: "v"(loff), "v"(ga) : "memory");
}
__device__ __forceinline__ void wait_async0() {
    asm volatile("s_wait_asynccnt 0x0" ::: "memory");
}

// ---------- prep: fp32 -> bf16 ----------
__global__ void cvt_x_kernel(const float* __restrict__ x, bf16* __restrict__ xb, int total) {
    int i = blockIdx.x * blockDim.x + threadIdx.x;
    if (i < total) xb[i] = (bf16)x[i];
}

// wT[out][in] = (bf16) w[in][out]  for 4 weights, selected by blockIdx.y
__global__ void transpose_w_kernel(const float* __restrict__ s0, const float* __restrict__ s1,
                                   const float* __restrict__ s2, const float* __restrict__ s3,
                                   bf16* __restrict__ d0, bf16* __restrict__ d1,
                                   bf16* __restrict__ d2, bf16* __restrict__ d3) {
    int z = blockIdx.y;
    const float* s = (z == 0) ? s0 : (z == 1) ? s1 : (z == 2) ? s2 : s3;
    bf16*        d = (z == 0) ? d0 : (z == 1) ? d1 : (z == 2) ? d2 : d3;
    int idx = blockIdx.x * blockDim.x + threadIdx.x; // 0 .. C*C-1
    int o = idx >> 10, i = idx & (Cc - 1);
    d[(size_t)o * Cc + i] = (bf16)s[(size_t)i * Cc + o];
}

// ---------- QKV projection GEMM ----------
// Y = xb @ W ; each wave owns a 16x64 output tile. blockIdx.z selects Q/K/V.
// Q -> [B,H,N,D] scaled ; K -> [B,H,N,D] ; V -> transposed [B,H,D,N].
__global__ __launch_bounds__(256)
void qkv_gemm_kernel(const bf16* __restrict__ xb,
                     const bf16* __restrict__ wqt, const bf16* __restrict__ wkt,
                     const bf16* __restrict__ wvt,
                     bf16* __restrict__ qo, bf16* __restrict__ ko, bf16* __restrict__ vo) {
    const int tid  = threadIdx.x;
    const int wave = tid >> 5, lane = tid & 31;
    const int half = lane >> 4, n16 = lane & 15;
    const int z = blockIdx.z;
    const bf16* wt = (z == 0) ? wqt : (z == 1) ? wkt : wvt;

    const int rowbase = blockIdx.x * 128 + wave * 16;
    const int colbase = blockIdx.y * 64;
    const bf16* arow = xb + (size_t)(rowbase + n16) * Cc;

    v8f acc[4] = {ZERO8, ZERO8, ZERO8, ZERO8};
    for (int kc = 0; kc < Cc; kc += 32) {
        const bf16* ap = arow + kc + 16 * half;
        v16bf a = make_frag(ap, ap + 8);
        #pragma unroll
        for (int f = 0; f < 4; ++f) {
            const bf16* wp = wt + (size_t)(colbase + f * 16 + n16) * Cc + kc + 16 * half;
            acc[f] = wmma_bf16(a, make_frag(wp, wp + 8), acc[f]);
        }
    }
    #pragma unroll
    for (int f = 0; f < 4; ++f) {
        const int colg = colbase + f * 16 + n16;
        const int h = colg >> 6, d = colg & (Dc - 1);
        #pragma unroll
        for (int g = 0; g < 8; ++g) {
            const int rowg = rowbase + g + 8 * half;
            const int b = rowg >> 11, nseq = rowg & (Nc - 1);
            const size_t bh = (size_t)(b * Hc + h);
            float v = acc[f][g];
            if (z == 0)      qo[(bh * Nc + nseq) * Dc + d] = (bf16)(v * SCALE);
            else if (z == 1) ko[(bh * Nc + nseq) * Dc + d] = (bf16)v;
            else             vo[(bh * Dc + d) * Nc + nseq] = (bf16)v;
        }
    }
}

// ---------- flash attention core ----------
// Per wave: 16 query rows with online softmax. All 8 waves of a block share
// (b,h), so each 32-key K/V tile is DMA'd into LDS ONCE per block with
// global_load_async_to_lds_b128, double-buffered to overlap with WMMAs.
// blockIdx.y = b*H + h ; blockIdx.x = tile of 128 rows (8 waves x 16).
constexpr int KPITCH = 72; // 64 + 8 bf16 pad: conflict-free b128 row reads
constexpr int VPITCH = 40; // 32 + 8 bf16 pad

__global__ __launch_bounds__(256)
void flash_attn_kernel(const bf16* __restrict__ qb, const bf16* __restrict__ kb,
                       const bf16* __restrict__ vtb, bf16* __restrict__ attn) {
    __shared__ bf16 kT[2][32 * KPITCH];  // K tile: 32 keys x 64 d
    __shared__ bf16 vT[2][64 * VPITCH];  // V^T tile: 64 d x 32 keys
    __shared__ bf16 lds_p[8][16 * 40];   // per-wave P staging (C/D -> A layout)

    const int tid  = threadIdx.x;
    const int wave = tid >> 5, lane = tid & 31;
    const int half = lane >> 4, n16 = lane & 15;

    const int bh = blockIdx.y;
    const int b = bh >> 4, h = bh & (Hc - 1);
    const int qbase = blockIdx.x * 128 + wave * 16;

    const bf16* qrow  = qb  + ((size_t)bh * Nc + qbase + n16) * Dc;
    const bf16* kbase = kb  + (size_t)bh * Nc * Dc;
    const bf16* vbase = vtb + (size_t)bh * Dc * Nc;
    bf16* myp = &lds_p[wave][0];

    // per-thread async chunk assignment (256 threads cover each tile)
    const int krow = tid >> 3, kck = (tid & 7) * 8;  // 32 rows x 8 chunks of 16B
    const int vrow = tid >> 2, vck = (tid & 3) * 8;  // 64 rows x 4 chunks of 16B

    // Q A-fragments: d-chunks [0,32) and [32,64)
    v16bf qa0 = make_frag(qrow + 16 * half,      qrow + 16 * half + 8);
    v16bf qa1 = make_frag(qrow + 32 + 16 * half, qrow + 32 + 16 * half + 8);

    float mrun[8], lrun[8], alpha[8];
    #pragma unroll
    for (int g = 0; g < 8; ++g) { mrun[g] = -__builtin_inff(); lrun[g] = 0.f; }
    v8f oacc[4] = {ZERO8, ZERO8, ZERO8, ZERO8};

    // prologue: DMA tile 0 into buffer 0
    async_ld_b128(kbase + (size_t)krow * Dc + kck,        &kT[0][krow * KPITCH + kck]);
    async_ld_b128(vbase + (size_t)vrow * Nc + vck,        &vT[0][vrow * VPITCH + vck]);

    for (int jb = 0; jb < Nc; jb += 32) {
        const int ib = (jb >> 5) & 1, nb = ib ^ 1;
        wait_async0();      // my async DMAs for buffer ib are done
        __syncthreads();    // everyone's DMAs visible; buffer nb fully consumed
        if (jb + 32 < Nc) { // prefetch next tile while we compute this one
            async_ld_b128(kbase + (size_t)(jb + 32 + krow) * Dc + kck, &kT[nb][krow * KPITCH + kck]);
            async_ld_b128(vbase + (size_t)vrow * Nc + jb + 32 + vck,   &vT[nb][vrow * VPITCH + vck]);
        }
        // S = Q * K^T  (16 queries x 32 keys), two 16-col halves, K from LDS
        v8f c0 = ZERO8, c1 = ZERO8;
        {
            const bf16* kp = &kT[ib][(n16) * KPITCH + 16 * half];
            c0 = wmma_bf16(qa0, make_frag(kp,      kp + 8),  c0);
            c0 = wmma_bf16(qa1, make_frag(kp + 32, kp + 40), c0);
        }
        {
            const bf16* kp = &kT[ib][(16 + n16) * KPITCH + 16 * half];
            c1 = wmma_bf16(qa0, make_frag(kp,      kp + 8),  c1);
            c1 = wmma_bf16(qa1, make_frag(kp + 32, kp + 40), c1);
        }
        // Online softmax; rows r = g + 8*half live across 16 lanes (col = n16)
        #pragma unroll
        for (int g = 0; g < 8; ++g) {
            float s0 = c0[g], s1 = c1[g];
            float mt = rmax16(fmaxf(s0, s1));
            float mn = fmaxf(mrun[g], mt);
            float a  = __expf(mrun[g] - mn);
            alpha[g] = a; mrun[g] = mn;
            float p0 = __expf(s0 - mn);
            float p1 = __expf(s1 - mn);
            lrun[g] = lrun[g] * a + rsum16(p0 + p1);
            const int r = g + 8 * half;
            myp[r * 40 + n16]      = (bf16)p0;   // cols 0..15
            myp[r * 40 + 16 + n16] = (bf16)p1;   // cols 16..31
        }
        // Intra-wave LDS exchange fence (DS ops are in-order per wave).
        asm volatile("s_wait_dscnt 0x0" ::: "memory");
        // Reload P in A-fragment layout: row = n16, kb = 16*half
        const bf16* pr = &myp[n16 * 40 + 16 * half];
        v16bf pf = make_frag(pr, pr + 8);
        // O = alpha*O + P * V   (V^T rows from LDS, contiguous in keys)
        #pragma unroll
        for (int f = 0; f < 4; ++f) {
            #pragma unroll
            for (int g = 0; g < 8; ++g) oacc[f][g] *= alpha[g];
            const bf16* vp = &vT[ib][(f * 16 + n16) * VPITCH + 16 * half];
            oacc[f] = wmma_bf16(pf, make_frag(vp, vp + 8), oacc[f]);
        }
    }
    // Normalize and scatter to attn[B,N,C] (bf16)
    #pragma unroll
    for (int f = 0; f < 4; ++f) {
        #pragma unroll
        for (int g = 0; g < 8; ++g) {
            const int r = g + 8 * half;
            attn[((size_t)b * Nc + qbase + r) * Cc + h * Dc + f * 16 + n16] =
                (bf16)(oacc[f][g] / lrun[g]);
        }
    }
}

// ---------- output projection: out = attn @ wo + bo (fp32 out) ----------
__global__ __launch_bounds__(256)
void out_proj_kernel(const bf16* __restrict__ attn, const bf16* __restrict__ wot,
                     const float* __restrict__ bo, float* __restrict__ out) {
    const int tid  = threadIdx.x;
    const int wave = tid >> 5, lane = tid & 31;
    const int half = lane >> 4, n16 = lane & 15;
    const int rowbase = blockIdx.x * 128 + wave * 16;
    const int colbase = blockIdx.y * 64;
    const bf16* arow = attn + (size_t)(rowbase + n16) * Cc;

    v8f acc[4] = {ZERO8, ZERO8, ZERO8, ZERO8};
    for (int kc = 0; kc < Cc; kc += 32) {
        const bf16* ap = arow + kc + 16 * half;
        v16bf a = make_frag(ap, ap + 8);
        #pragma unroll
        for (int f = 0; f < 4; ++f) {
            const bf16* wp = wot + (size_t)(colbase + f * 16 + n16) * Cc + kc + 16 * half;
            acc[f] = wmma_bf16(a, make_frag(wp, wp + 8), acc[f]);
        }
    }
    #pragma unroll
    for (int f = 0; f < 4; ++f) {
        const int colg = colbase + f * 16 + n16;
        const float bias = bo[colg];
        #pragma unroll
        for (int g = 0; g < 8; ++g) {
            const int rowg = rowbase + g + 8 * half;
            out[(size_t)rowg * Cc + colg] = acc[f][g] + bias;
        }
    }
}

// ---------- host-side orchestration ----------
extern "C" void kernel_launch(void* const* d_in, const int* in_sizes, int n_in,
                              void* d_out, int out_size, void* d_ws, size_t ws_size,
                              hipStream_t stream) {
    const float* x  = (const float*)d_in[0];
    const float* wq = (const float*)d_in[1];
    const float* wk = (const float*)d_in[2];
    const float* wv = (const float*)d_in[3];
    const float* wo = (const float*)d_in[4];
    const float* bo = (const float*)d_in[5];
    float* out = (float*)d_out;

    // workspace carve (bytes)
    char* ws = (char*)d_ws;
    const size_t szX = (size_t)Bc * Nc * Cc * sizeof(bf16);   // 16 MB
    const size_t szW = (size_t)Cc * Cc * sizeof(bf16);        // 2 MB
    bf16* xb   = (bf16*)(ws);                         // also reused as attn buffer
    bf16* wqt  = (bf16*)(ws + szX);
    bf16* wkt  = (bf16*)(ws + szX + szW);
    bf16* wvt  = (bf16*)(ws + szX + 2 * szW);
    bf16* wot  = (bf16*)(ws + szX + 3 * szW);
    bf16* qbuf = (bf16*)(ws + szX + 4 * szW);
    bf16* kbuf = (bf16*)(ws + szX + 4 * szW + szX);
    bf16* vtbf = (bf16*)(ws + szX + 4 * szW + 2 * szX);
    bf16* attn = xb; // xb is dead after qkv_gemm

    const int totalX = Bc * Nc * Cc; // 8388608
    cvt_x_kernel<<<dim3(totalX / 256), dim3(256), 0, stream>>>(x, xb, totalX);
    transpose_w_kernel<<<dim3(Cc * Cc / 256, 4), dim3(256), 0, stream>>>(
        wq, wk, wv, wo, wqt, wkt, wvt, wot);

    // M=8192 rows / 128 per block, Nout=1024 / 64 per block, z: q/k/v
    qkv_gemm_kernel<<<dim3(Bc * Nc / 128, Cc / 64, 3), dim3(256), 0, stream>>>(
        xb, wqt, wkt, wvt, qbuf, kbuf, vtbf);

    flash_attn_kernel<<<dim3(Nc / 128, Bc * Hc), dim3(256), 0, stream>>>(
        qbuf, kbuf, vtbf, attn);

    out_proj_kernel<<<dim3(Bc * Nc / 128, Cc / 64), dim3(256), 0, stream>>>(
        attn, wot, bo, out);
}
